// DirectMPIGO_8830452761402
// MI455X (gfx1250) — matrix operand order repaired and checked by reading, not compiled, not run on hardware
//
#include <hip/hip_runtime.h>
#include <hip/hip_bf16.h>

// ---------------------------------------------------------------------------
// DirectMPIGO renderer for MI455X (gfx1250, wave32).
// Block = 1 ray (128 samples = GEMM M-tile of 128). 4 waves x 2 sub-tiles.
// MLP (66->128->128->3) on v_wmma_f32_16x16x32_f16 with fp32 accumulation.
// ---------------------------------------------------------------------------

typedef __attribute__((ext_vector_type(16))) _Float16 v16h;
typedef __attribute__((ext_vector_type(8)))  _Float16 v8h;
typedef __attribute__((ext_vector_type(8)))  float    v8f;

#define NRAYS     8192
#define NSAMP     128
#define FSTRIDE   144      // halves per LDS row = 288B (32B-aligned, 72-dword stride)
#define K0PAD     96       // feature K padded 66 -> 96 (3 chunks of K=32)
#define INTERVALv 2.0f     // 256/128
#define BGv       1.0f

// workspace layout (in halves)
#define W0T_OFF   0                 // 128 x 96   (w0^T, K-padded)
#define W1T_OFF   12288             // 128 x 128  (w1^T)
#define W2T_OFF   28672             // 16  x 128  (w2^T, N-padded 3->16)
#define WT_TOTAL  30720

__device__ __forceinline__ v8f wmma16(v16h a, v16h b, v8f c) {
    return __builtin_amdgcn_wmma_f32_16x16x32_f16(
        /*neg_a=*/false, a, /*neg_b=*/false, b,
        /*c_mod=*/(short)0, c, /*reuse_a=*/false, /*reuse_b=*/false);
}

// A fragment (16x32 f16): lane L holds row M = L%16.
// lanes 0-15: K = base+[0..15]; lanes 16-31: K = base+[16..31].
__device__ __forceinline__ v16h loadA(const _Float16* buf, int mtile, int kc,
                                      int nlane, bool hi) {
    const _Float16* p = buf + (mtile * 16 + nlane) * FSTRIDE + kc * 32 + (hi ? 16 : 0);
    v8h lo = *(const v8h*)(p);
    v8h ho = *(const v8h*)(p + 8);
    v16h a;
#pragma unroll
    for (int i = 0; i < 8; ++i) { a[i] = lo[i]; a[8 + i] = ho[i]; }
    return a;
}

// B fragment (32x16 f16): lane L holds column N = L%16,
// K = kc*32 + (L<16 ? [0..15] : [16..31]) -- 16 contiguous halves of w^T[n][:].
__device__ __forceinline__ v16h loadB(const _Float16* wT, int K, int n, int kc,
                                      bool hi) {
    return *(const v16h*)(wT + (size_t)n * K + kc * 32 + (hi ? 16 : 0));
}

// ---------------------------------------------------------------------------
// Prep: transpose + f16-convert + pad the MLP weights into d_ws.
// ---------------------------------------------------------------------------
__global__ void prep_weights(const float* __restrict__ w0,
                             const float* __restrict__ w1,
                             const float* __restrict__ w2,
                             _Float16* __restrict__ wT) {
    int i = blockIdx.x * blockDim.x + threadIdx.x;
    if (i < 12288) {                       // w0T[n][k], k padded 66->96
        int n = i / 96, k = i % 96;
        wT[W0T_OFF + i] = (_Float16)(k < 66 ? w0[k * 128 + n] : 0.f);
    } else if (i < 12288 + 16384) {        // w1T[n][k]
        int j = i - 12288;
        int n = j >> 7, k = j & 127;
        wT[W1T_OFF + j] = (_Float16)w1[k * 128 + n];
    } else if (i < WT_TOTAL) {             // w2T[n][k], n padded 3->16
        int j = i - 28672;
        int n = j >> 7, k = j & 127;
        wT[W2T_OFF + j] = (_Float16)(n < 3 ? w2[k * 3 + n] : 0.f);
    }
}

// ---------------------------------------------------------------------------
// Main fused kernel: 1 block per ray, 128 threads (4 waves).
// ---------------------------------------------------------------------------
__global__ __launch_bounds__(128) void mpi_render(
    const float* __restrict__ rays_o, const float* __restrict__ rays_d,
    const float* __restrict__ dgrid,  const float* __restrict__ k0grid,
    const float* __restrict__ act_shift,
    const _Float16* __restrict__ w0T, const _Float16* __restrict__ w1T,
    const _Float16* __restrict__ w2T,
    const float* __restrict__ b0, const float* __restrict__ b1,
    const float* __restrict__ b2,
    float* __restrict__ out) {
    extern __shared__ char smem[];
    _Float16* bufA = (_Float16*)smem;                        // feat / h1
    _Float16* bufB = (_Float16*)(smem + 128 * FSTRIDE * 2);  // h0
    float* sAlpha  = (float*)(smem + 2 * 128 * FSTRIDE * 2);
    float* sCum    = sAlpha + 128;
    float* sW      = sCum + 128;
    float* sOut    = sW + 128;     // 4 floats

    const int r = blockIdx.x;
    const int s = threadIdx.x;                 // sample index == GEMM row
    const int lane  = threadIdx.x & 31;
    const int wv    = threadIdx.x >> 5;
    const bool hi   = lane >= 16;
    const int nlane = lane & 15;

    // ---------------- Phase 1: per-sample geometry + features ----------------
    const float ox = rays_o[r * 3 + 0], oy = rays_o[r * 3 + 1], oz = rays_o[r * 3 + 2];
    const float dx = rays_d[r * 3 + 0], dy = rays_d[r * 3 + 1], dz = rays_d[r * 3 + 2];
    const float t  = (float)s * (1.0f / 127.0f);
    const float px = ox + dx * t, py = oy + dy * t, pz = oz + dz * t;

    // trilerp coords: grid (256,256,128), xyz in ([-1,1],[-1,1],[0,1])
    float ux = fminf(fmaxf((px + 1.f) * 0.5f * 255.f, 0.f), 255.f);
    float uy = fminf(fmaxf((py + 1.f) * 0.5f * 255.f, 0.f), 255.f);
    float uz = fminf(fmaxf(pz * 127.f, 0.f), 127.f);
    int ix0 = min((int)floorf(ux), 254);
    int iy0 = min((int)floorf(uy), 254);
    int iz0 = min((int)floorf(uz), 126);
    float fx = ux - (float)ix0, fy = uy - (float)iy0, fz = uz - (float)iz0;
    float wxc[2] = {1.f - fx, fx}, wyc[2] = {1.f - fy, fy}, wzc[2] = {1.f - fz, fz};

    float dens = 0.f;
    float k0a[12];
#pragma unroll
    for (int c = 0; c < 12; ++c) k0a[c] = 0.f;

#pragma unroll
    for (int a = 0; a < 2; ++a)
#pragma unroll
        for (int b = 0; b < 2; ++b) {
            const size_t base2 = ((size_t)(ix0 + a) * 256 + (iy0 + b)) * 128;
#pragma unroll
            for (int c = 0; c < 2; ++c) {
                const float w   = wxc[a] * wyc[b] * wzc[c];
                const size_t id = base2 + (iz0 + c);
                dens += w * dgrid[id];
                const float4* kp = (const float4*)(k0grid + id * 12);
                float4 k0v = kp[0], k1v = kp[1], k2v = kp[2];
                k0a[0] += w * k0v.x; k0a[1]  += w * k0v.y; k0a[2]  += w * k0v.z; k0a[3]  += w * k0v.w;
                k0a[4] += w * k1v.x; k0a[5]  += w * k1v.y; k0a[6]  += w * k1v.z; k0a[7]  += w * k1v.w;
                k0a[8] += w * k2v.x; k0a[9]  += w * k2v.y; k0a[10] += w * k2v.z; k0a[11] += w * k2v.w;
            }
        }

    // act_shift 1D lerp along z (same uz/iz0 mapping, depth 128)
    const float shift = wzc[0] * act_shift[iz0] + wzc[1] * act_shift[iz0 + 1];
    const float sd = dens + shift;
    const float sp = sd > 15.f ? sd : log1pf(__expf(sd));
    const float alpha = 1.f - __expf(-sp * INTERVALv);
    sAlpha[s] = alpha;
    sCum[s]   = 1.f - alpha;

    // feature row: [k0(12) | x,pe_sin(12),pe_cos(12) | v,pe_sin(12),pe_cos(12) | pad]
    const float rn = rsqrtf(dx * dx + dy * dy + dz * dz);
    const float pxyz[3] = {px, py, pz};
    const float v3[3]   = {dx * rn, dy * rn, dz * rn};
    const float freqs[4] = {1.f, 2.f, 4.f, 8.f};

    _Float16* fr = bufA + s * FSTRIDE;
#pragma unroll
    for (int c = 0; c < 12; ++c) fr[c] = (_Float16)k0a[c];
    int o = 12;
#pragma unroll
    for (int a = 0; a < 3; ++a) fr[o++] = (_Float16)pxyz[a];
#pragma unroll
    for (int a = 0; a < 3; ++a)
#pragma unroll
        for (int j = 0; j < 4; ++j) fr[o++] = (_Float16)__sinf(pxyz[a] * freqs[j]);
#pragma unroll
    for (int a = 0; a < 3; ++a)
#pragma unroll
        for (int j = 0; j < 4; ++j) fr[o++] = (_Float16)__cosf(pxyz[a] * freqs[j]);
#pragma unroll
    for (int a = 0; a < 3; ++a) fr[o++] = (_Float16)v3[a];
#pragma unroll
    for (int a = 0; a < 3; ++a)
#pragma unroll
        for (int j = 0; j < 4; ++j) fr[o++] = (_Float16)__sinf(v3[a] * freqs[j]);
#pragma unroll
    for (int a = 0; a < 3; ++a)
#pragma unroll
        for (int j = 0; j < 4; ++j) fr[o++] = (_Float16)__cosf(v3[a] * freqs[j]);
    for (; o < K0PAD; ++o) fr[o] = (_Float16)0.f;

    // ---------------- Phase 2: transmittance scan (Hillis-Steele) ------------
    __syncthreads();
    for (int off = 1; off < NSAMP; off <<= 1) {
        float self  = sCum[s];
        float other = (s >= off) ? sCum[s - off] : 1.f;
        __syncthreads();
        sCum[s] = self * other;
        __syncthreads();
    }
    sW[s] = sAlpha[s] * (s == 0 ? 1.f : sCum[s - 1]);
    if (s < 3) sOut[s] = 0.f;
    __syncthreads();

    // ---------------- Phase 3: MLP on WMMA ------------------------------------
    // Layer 0: feat(bufA, K=96) @ w0 -> relu -> bufB (128 cols)
    for (int mt = wv * 2; mt < wv * 2 + 2; ++mt) {
        v16h a0 = loadA(bufA, mt, 0, nlane, hi);
        v16h a1 = loadA(bufA, mt, 1, nlane, hi);
        v16h a2 = loadA(bufA, mt, 2, nlane, hi);
#pragma unroll
        for (int nt = 0; nt < 8; ++nt) {
            const int n = nt * 16 + nlane;
            const float bv = b0[n];
            v8f acc = {bv, bv, bv, bv, bv, bv, bv, bv};
            acc = wmma16(a0, loadB(w0T, 96, n, 0, hi), acc);
            acc = wmma16(a1, loadB(w0T, 96, n, 1, hi), acc);
            acc = wmma16(a2, loadB(w0T, 96, n, 2, hi), acc);
#pragma unroll
            for (int g = 0; g < 8; ++g) {
                const int m = mt * 16 + (hi ? 8 : 0) + g;
                bufB[m * FSTRIDE + n] = (_Float16)fmaxf(acc[g], 0.f);
            }
        }
    }
    __syncthreads();

    // Layer 1: h0(bufB, K=128) @ w1 -> relu -> bufA
    for (int mt = wv * 2; mt < wv * 2 + 2; ++mt) {
        v16h a0 = loadA(bufB, mt, 0, nlane, hi);
        v16h a1 = loadA(bufB, mt, 1, nlane, hi);
        v16h a2 = loadA(bufB, mt, 2, nlane, hi);
        v16h a3 = loadA(bufB, mt, 3, nlane, hi);
#pragma unroll
        for (int nt = 0; nt < 8; ++nt) {
            const int n = nt * 16 + nlane;
            const float bv = b1[n];
            v8f acc = {bv, bv, bv, bv, bv, bv, bv, bv};
            acc = wmma16(a0, loadB(w1T, 128, n, 0, hi), acc);
            acc = wmma16(a1, loadB(w1T, 128, n, 1, hi), acc);
            acc = wmma16(a2, loadB(w1T, 128, n, 2, hi), acc);
            acc = wmma16(a3, loadB(w1T, 128, n, 3, hi), acc);
#pragma unroll
            for (int g = 0; g < 8; ++g) {
                const int m = mt * 16 + (hi ? 8 : 0) + g;
                bufA[m * FSTRIDE + n] = (_Float16)fmaxf(acc[g], 0.f);
            }
        }
    }
    __syncthreads();

    // Layer 2: h1(bufA, K=128) @ w2 (N padded to 16) -> sigmoid -> weighted sum
    for (int mt = wv * 2; mt < wv * 2 + 2; ++mt) {
        v16h a0 = loadA(bufA, mt, 0, nlane, hi);
        v16h a1 = loadA(bufA, mt, 1, nlane, hi);
        v16h a2 = loadA(bufA, mt, 2, nlane, hi);
        v16h a3 = loadA(bufA, mt, 3, nlane, hi);
        const float bv = (nlane < 3) ? b2[nlane] : 0.f;
        v8f acc = {bv, bv, bv, bv, bv, bv, bv, bv};
        acc = wmma16(a0, loadB(w2T, 128, nlane, 0, hi), acc);
        acc = wmma16(a1, loadB(w2T, 128, nlane, 1, hi), acc);
        acc = wmma16(a2, loadB(w2T, 128, nlane, 2, hi), acc);
        acc = wmma16(a3, loadB(w2T, 128, nlane, 3, hi), acc);
        if (nlane < 3) {
            float partial = 0.f;
#pragma unroll
            for (int g = 0; g < 8; ++g) {
                const int m = mt * 16 + (hi ? 8 : 0) + g;
                const float rgb = 1.f / (1.f + __expf(-acc[g]));
                partial += sW[m] * rgb;
            }
            atomicAdd(&sOut[nlane], partial);   // ds_add_f32
        }
    }
    __syncthreads();

    // ---------------- Phase 4: write out --------------------------------------
    if (s < 3) out[r * 3 + s] = sOut[s] + sCum[NSAMP - 1] * BGv;
}

// ---------------------------------------------------------------------------
extern "C" void kernel_launch(void* const* d_in, const int* in_sizes, int n_in,
                              void* d_out, int out_size, void* d_ws, size_t ws_size,
                              hipStream_t stream) {
    const float* rays_o    = (const float*)d_in[0];
    const float* rays_d    = (const float*)d_in[1];
    const float* dgrid     = (const float*)d_in[2];
    const float* k0grid    = (const float*)d_in[3];
    const float* act_shift = (const float*)d_in[4];
    const float* w0        = (const float*)d_in[5];
    const float* b0        = (const float*)d_in[6];
    const float* w1        = (const float*)d_in[7];
    const float* b1        = (const float*)d_in[8];
    const float* w2        = (const float*)d_in[9];
    const float* b2        = (const float*)d_in[10];
    float* out             = (float*)d_out;
    _Float16* wT           = (_Float16*)d_ws;

    prep_weights<<<(WT_TOTAL + 255) / 256, 256, 0, stream>>>(w0, w1, w2, wT);

    const size_t smem = 2 * 128 * FSTRIDE * sizeof(_Float16)   // two activation buffers
                      + (3 * 128 + 4) * sizeof(float);          // alpha/cum/weights + accum
    mpi_render<<<NRAYS, 128, smem, stream>>>(
        rays_o, rays_d, dgrid, k0grid, act_shift,
        wT + W0T_OFF, wT + W1T_OFF, wT + W2T_OFF,
        b0, b1, b2, out);
}
